// Wave_Conv_ViT_38208029065988
// MI455X (gfx1250) — compile-verified, matrix operand order checked by reading, not run on hardware
//
#include <hip/hip_runtime.h>
#include <hip/hip_bf16.h>
#include <math.h>

// ---------------- problem constants ----------------
#define DIMC   256
#define NHEAD  8
#define WSZ    8
#define BATCH  16
#define IMH    64
#define IMW    64
#define LTOK   (IMH*IMW)          // 4096
#define MTOK   (BATCH*LTOK)       // 65536 rows (windowed or not)
#define NWIN   (BATCH*64)         // 1024 windows
#define MW16   (NWIN*16)          // 16384 half-window rows

typedef __hip_bfloat16 bf16_t;
typedef __attribute__((ext_vector_type(16))) __bf16        v16bf;
typedef __attribute__((ext_vector_type(8)))  float         v8f;
typedef __attribute__((ext_vector_type(4)))  unsigned int  u32x4;

union Frag32 { u32x4 u[2]; v16bf v; };

// windowed row r (ws=8) -> (batch b, token l) in the flat (B,L,C) tensor
__device__ __forceinline__ void win_map(int r, int& b, int& l) {
  int t = r >> 6;                 // window id = (b*8+hb)*8+wb
  int s = r & 63;                 // in-window token = i*8+j
  int wb = t & 7, hb = (t >> 3) & 7;
  b = t >> 6;
  int i = s >> 3, j = s & 7;
  l = (hb * 8 + i) * IMW + (wb * 8 + j);
}

// ---------------- WMMA GEMM: C = act(A @ B^T + bias) ----------------
// A: M x K bf16 row-major.  Bt: N x K bf16 (weights pre-transposed so each
// lane's K-chunks are contiguous b128 loads).  One wave -> 64x64 tile:
// 4x4 accumulator frags, 16 WMMAs per 16 b128 loads per K-step of 32.
// MODE: 0 store f32, 1 store bf16, 2 gelu(exact)->bf16, 3 leakyrelu->bf16
template<int MODE>
__global__ __launch_bounds__(256) void gemm_wmma_bf16(
    const bf16_t* __restrict__ A, const bf16_t* __restrict__ Bt,
    const float* __restrict__ bias, void* __restrict__ outp,
    int M, int N, int K)
{
  const int wave   = blockIdx.x * (blockDim.x >> 5) + (threadIdx.x >> 5);
  const int tilesN = N >> 6;
  const int tilesM = M >> 6;
  if (wave >= tilesM * tilesN) return;          // wave-uniform
  const int tm = (wave / tilesN) << 6;
  const int tn = (wave % tilesN) << 6;
  const int lane = threadIdx.x & 31;
  const int hf   = lane >> 4;                   // K-half selector (ISA layout)
  const int lr   = lane & 15;

  v8f acc[4][4] = {};

  const bf16_t* aP[4];
  const bf16_t* bP[4];
  #pragma unroll
  for (int i = 0; i < 4; ++i) {
    aP[i] = A  + (size_t)(tm + i * 16 + lr) * K + hf * 8;
    bP[i] = Bt + (size_t)(tn + i * 16 + lr) * K + hf * 8;
  }

  for (int k0 = 0; k0 < K; k0 += 32) {
    Frag32 af[4], bfr[4];
    #pragma unroll
    for (int i = 0; i < 4; ++i) {
      const u32x4* pa = (const u32x4*)(aP[i] + k0);
      // chunk0 = K[hf*8 .. hf*8+8), chunk1 = K[16+hf*8 .. 16+hf*8+8)
      af[i].u[0] = pa[0]; af[i].u[1] = pa[2];
    }
    #pragma unroll
    for (int i = 0; i < 4; ++i) {
      const u32x4* pb = (const u32x4*)(bP[i] + k0);
      bfr[i].u[0] = pb[0]; bfr[i].u[1] = pb[2];
    }
    if (k0 + 32 < K) {                          // -> global_prefetch_b8
      __builtin_prefetch((const void*)(aP[0] + k0 + 32), 0, 3);
      __builtin_prefetch((const void*)(bP[0] + k0 + 32), 0, 3);
    }
    #pragma unroll
    for (int mi = 0; mi < 4; ++mi) {
      #pragma unroll
      for (int ni = 0; ni < 4; ++ni) {
        acc[mi][ni] = __builtin_amdgcn_wmma_f32_16x16x32_bf16(
            false, af[mi].v, false, bfr[ni].v, (short)0, acc[mi][ni], false, false);
      }
    }
  }

  #pragma unroll
  for (int mi = 0; mi < 4; ++mi) {
    #pragma unroll
    for (int ni = 0; ni < 4; ++ni) {
      const int n  = tn + ni * 16 + lr;
      const float bv = bias ? bias[n] : 0.0f;
      #pragma unroll
      for (int r = 0; r < 8; ++r) {
        const int m = tm + mi * 16 + hf * 8 + r;   // C/D layout: vgpr r, half->+8
        float v = acc[mi][ni][r] + bv;
        const size_t o = (size_t)m * N + n;
        if constexpr (MODE == 0) {
          ((float*)outp)[o] = v;
        } else if constexpr (MODE == 1) {
          ((bf16_t*)outp)[o] = __float2bfloat16(v);
        } else if constexpr (MODE == 2) {
          float g = 0.5f * v * (1.0f + erff(v * 0.7071067811865475f));
          ((bf16_t*)outp)[o] = __float2bfloat16(g);
        } else {
          float g = v > 0.0f ? v : 0.2f * v;
          ((bf16_t*)outp)[o] = __float2bfloat16(g);
        }
      }
    }
  }
}

// ---------------- weight transpose + f32 -> bf16 ----------------
__global__ void wconv_t_kernel(const float* __restrict__ src, bf16_t* __restrict__ dst,
                               int K, int N)
{
  size_t idx = (size_t)blockIdx.x * blockDim.x + threadIdx.x;
  if (idx >= (size_t)K * N) return;
  int n = (int)(idx % N);
  int k = (int)(idx / N);
  dst[(size_t)n * K + k] = __float2bfloat16(src[idx]);
}

// ---------------- depthwise 3x3 conv (channel-last, SAME) ----------------
// ACT 0: out = in + conv(in)   (CPE residual)
// ACT 1: out = leakyrelu(conv(in), 0.2)
template<int ACT>
__global__ void dwconv_kernel(const float* __restrict__ in, const float* __restrict__ w,
                              const float* __restrict__ bias, float* __restrict__ out,
                              int B, int Hh, int Ww)
{
  size_t idx = (size_t)blockIdx.x * blockDim.x + threadIdx.x;
  size_t total = (size_t)B * Hh * Ww * DIMC;
  if (idx >= total) return;
  int c = (int)(idx & (DIMC - 1));
  size_t p = idx >> 8;
  int x = (int)(p % Ww); p /= Ww;
  int y = (int)(p % Hh);
  int b = (int)(p / Hh);
  float acc = bias[c];
  #pragma unroll
  for (int ky = 0; ky < 3; ++ky) {
    int yy = y + ky - 1;
    if (yy < 0 || yy >= Hh) continue;
    #pragma unroll
    for (int kx = 0; kx < 3; ++kx) {
      int xx = x + kx - 1;
      if (xx < 0 || xx >= Ww) continue;
      acc += in[(((size_t)b * Hh + yy) * Ww + xx) * DIMC + c] * w[c * 9 + ky * 3 + kx];
    }
  }
  if (ACT == 0) out[idx] = in[idx] + acc;
  else          out[idx] = acc > 0.0f ? acc : 0.2f * acc;
}

// ---------------- window-gather x (f32) -> xw (bf16 GEMM-A) ----------------
__global__ void gather_win_kernel(const float* __restrict__ x1, bf16_t* __restrict__ xw)
{
  size_t idx = (size_t)blockIdx.x * blockDim.x + threadIdx.x;
  if (idx >= (size_t)MTOK * DIMC) return;
  int c = (int)(idx & (DIMC - 1));
  int r = (int)(idx >> 8);
  int b, l; win_map(r, b, l);
  xw[idx] = __float2bfloat16(x1[((size_t)b * LTOK + l) * DIMC + c]);
}

// ---------------- 2x2 Haar DWT -> f32 half-res planes + windowed bf16 ----------------
__global__ void dwt_kernel(const float* __restrict__ x1,
                           float* __restrict__ llF, float* __restrict__ lhF,
                           float* __restrict__ hlF, float* __restrict__ hhF,
                           bf16_t* __restrict__ ll_bf, bf16_t* __restrict__ hi_bf)
{
  size_t idx = (size_t)blockIdx.x * blockDim.x + threadIdx.x;
  if (idx >= (size_t)BATCH * 32 * 32 * DIMC) return;
  int c = (int)(idx & (DIMC - 1));
  size_t p = idx >> 8;
  int Q = (int)(p % 32); p /= 32;
  int P = (int)(p % 32);
  int b = (int)(p / 32);
  const size_t base = (((size_t)b * IMH + 2 * P) * IMW + 2 * Q) * DIMC + c;
  const float av = x1[base];
  const float bv = x1[base + DIMC];
  const float cv = x1[base + (size_t)IMW * DIMC];
  const float dv = x1[base + (size_t)IMW * DIMC + DIMC];
  const float ll = 0.5f * (av + bv + cv + dv);
  const float lh = 0.5f * (av + bv - cv - dv);
  const float hl = 0.5f * (av - bv + cv - dv);
  const float hh = 0.5f * (av - bv - cv + dv);
  const size_t fo = (((size_t)b * 32 + P) * 32 + Q) * DIMC + c;
  llF[fo] = ll; lhF[fo] = lh; hlF[fo] = hl; hhF[fo] = hh;
  const int b_ = (b * 8 + (P >> 2)) * 8 + (Q >> 2);
  const int s  = (P & 3) * 4 + (Q & 3);
  const size_t row = (size_t)b_ * 16 + s;
  ll_bf[row * DIMC + c] = __float2bfloat16(ll);
  hi_bf[row * 768 + c]        = __float2bfloat16(lh);
  hi_bf[row * 768 + 256 + c]  = __float2bfloat16(hl);
  hi_bf[row * 768 + 512 + c]  = __float2bfloat16(hh);
}

// ---------------- inverse DWT (to full res, channel-last) ----------------
__global__ void idwt_kernel(const float* __restrict__ ll, const float* __restrict__ lh,
                            const float* __restrict__ hl, const float* __restrict__ hh,
                            float* __restrict__ out)
{
  size_t idx = (size_t)blockIdx.x * blockDim.x + threadIdx.x;
  if (idx >= (size_t)MTOK * DIMC) return;
  int c = (int)(idx & (DIMC - 1));
  size_t p = idx >> 8;
  int x = (int)(p % IMW); p /= IMW;
  int y = (int)(p % IMH);
  int b = (int)(p / IMH);
  const size_t fo = (((size_t)b * 32 + (y >> 1)) * 32 + (x >> 1)) * DIMC + c;
  const float L = ll[fo], Lh = lh[fo], Hl = hl[fo], Hh = hh[fo];
  float v;
  if ((y & 1) == 0) v = (x & 1) == 0 ? (L + Lh + Hl + Hh) : (L + Lh - Hl - Hh);
  else              v = (x & 1) == 0 ? (L - Lh + Hl - Hh) : (L - Lh - Hl + Hh);
  out[idx] = 0.5f * v;
}

// ---------------- cosine window attention (64 q x 16 kv, dh=32) ----------------
__global__ __launch_bounds__(64) void attn_kernel(
    const float* __restrict__ Q, const float* __restrict__ KV,
    const float* __restrict__ ls, bf16_t* __restrict__ cat, int colOff)
{
  const int b_ = blockIdx.x;     // window
  const int h  = blockIdx.y;     // head
  __shared__ float Kn[16][32];
  __shared__ float Vs[16][32];
  __shared__ float sscale;
  const int t = threadIdx.x;
  if (t == 0) sscale = expf(fminf(ls[h], 4.6051701859880914f)); // ln(100)
  if (t < 16) {
    const float* kp = KV + ((size_t)(b_ * 16 + t)) * 512 + h * 32;
    const float* vp = kp + 256;
    float nrm = 0.0f; float kr[32];
    #pragma unroll
    for (int d = 0; d < 32; ++d) { kr[d] = kp[d]; nrm += kr[d] * kr[d]; }
    float inv = 1.0f / fmaxf(sqrtf(nrm), 1e-12f);
    #pragma unroll
    for (int d = 0; d < 32; ++d) { Kn[t][d] = kr[d] * inv; Vs[t][d] = vp[d]; }
  }
  __syncthreads();
  const float scale = sscale;
  const float* qp = Q + ((size_t)(b_ * 64 + t)) * DIMC + h * 32;
  float q[32]; float nrm = 0.0f;
  #pragma unroll
  for (int d = 0; d < 32; ++d) { q[d] = qp[d]; nrm += q[d] * q[d]; }
  const float qi = scale / fmaxf(sqrtf(nrm), 1e-12f);
  float s[16]; float mx = -1e30f;
  #pragma unroll
  for (int m = 0; m < 16; ++m) {
    float dot = 0.0f;
    #pragma unroll
    for (int d = 0; d < 32; ++d) dot += q[d] * Kn[m][d];
    s[m] = dot * qi;
    mx = fmaxf(mx, s[m]);
  }
  float sum = 0.0f;
  #pragma unroll
  for (int m = 0; m < 16; ++m) { s[m] = expf(s[m] - mx); sum += s[m]; }
  const float isum = 1.0f / sum;
  bf16_t* op = cat + ((size_t)(b_ * 64 + t)) * 768 + colOff;
  #pragma unroll
  for (int d = 0; d < 32; ++d) {
    float o = 0.0f;
    #pragma unroll
    for (int m = 0; m < 16; ++m) o += s[m] * Vs[m][d];
    op[d * NHEAD + h] = __float2bfloat16(o * isum);   // feature = d*nh + h
  }
}

// ---------------- scatter rst (full-res f32) into concat cols [512,768) ----------------
__global__ void scatter_rst_kernel(const float* __restrict__ rst, bf16_t* __restrict__ cat)
{
  size_t idx = (size_t)blockIdx.x * blockDim.x + threadIdx.x;
  if (idx >= (size_t)MTOK * DIMC) return;
  int c = (int)(idx & (DIMC - 1));
  int r = (int)(idx >> 8);
  int b, l; win_map(r, b, l);
  cat[(size_t)r * 768 + 512 + c] = __float2bfloat16(rst[((size_t)b * LTOK + l) * DIMC + c]);
}

// ---------------- LayerNorm(src row) * g + b + residual ----------------
// WINDOWED: src rows are in window order, output/residual in (B,L,C) order
template<int WINDOWED>
__global__ __launch_bounds__(256) void ln_add_kernel(
    const float* __restrict__ src, const float* __restrict__ res,
    const float* __restrict__ g, const float* __restrict__ bta,
    float* __restrict__ out)
{
  const int r = blockIdx.x;
  const int c = threadIdx.x;
  __shared__ float sm[256];
  __shared__ float sv[256];
  const float v = src[(size_t)r * DIMC + c];
  sm[c] = v; sv[c] = v * v;
  __syncthreads();
  for (int st = 128; st > 0; st >>= 1) {
    if (c < st) { sm[c] += sm[c + st]; sv[c] += sv[c + st]; }
    __syncthreads();
  }
  const float mean = sm[0] * (1.0f / DIMC);
  const float var  = sv[0] * (1.0f / DIMC) - mean * mean;
  const float ninv = rsqrtf(var + 1e-5f);
  size_t o;
  if (WINDOWED) { int b, l; win_map(r, b, l); o = ((size_t)b * LTOK + l) * DIMC + c; }
  else          { o = (size_t)r * DIMC + c; }
  out[o] = res[o] + (v - mean) * ninv * g[c] + bta[c];
}

// ---------------- f32 -> bf16 elementwise ----------------
__global__ void cvt_bf16_kernel(const float* __restrict__ src, bf16_t* __restrict__ dst,
                                size_t n)
{
  size_t idx = (size_t)blockIdx.x * blockDim.x + threadIdx.x;
  if (idx < n) dst[idx] = __float2bfloat16(src[idx]);
}

// ---------------- launcher ----------------
static void launch_gemm(int mode, const bf16_t* A, const bf16_t* Bt, const float* bias,
                        void* out, int M, int N, int K, hipStream_t s)
{
  int waves  = (M >> 6) * (N >> 6);             // 64x64 wave tiles
  int blocks = (waves + 7) / 8;                 // 8 waves (256 threads) per block
  switch (mode) {
    case 0: gemm_wmma_bf16<0><<<blocks, 256, 0, s>>>(A, Bt, bias, out, M, N, K); break;
    case 1: gemm_wmma_bf16<1><<<blocks, 256, 0, s>>>(A, Bt, bias, out, M, N, K); break;
    case 2: gemm_wmma_bf16<2><<<blocks, 256, 0, s>>>(A, Bt, bias, out, M, N, K); break;
    default: gemm_wmma_bf16<3><<<blocks, 256, 0, s>>>(A, Bt, bias, out, M, N, K); break;
  }
}

static inline unsigned grid1(size_t n) { return (unsigned)((n + 255) / 256); }

extern "C" void kernel_launch(void* const* d_in, const int* in_sizes, int n_in,
                              void* d_out, int out_size, void* d_ws, size_t ws_size,
                              hipStream_t stream)
{
  (void)in_sizes; (void)n_in; (void)out_size; (void)ws_size;
  // inputs (setup_inputs order)
  const float* x      = (const float*)d_in[0];
  const float* cpe0_w = (const float*)d_in[1];  const float* cpe0_b = (const float*)d_in[2];
  const float* cpe1_w = (const float*)d_in[3];  const float* cpe1_b = (const float*)d_in[4];
  const float* n1_g   = (const float*)d_in[5];  const float* n1_b   = (const float*)d_in[6];
  const float* n2_g   = (const float*)d_in[7];  const float* n2_b   = (const float*)d_in[8];
  const float* q_w    = (const float*)d_in[9];  const float* q_b    = (const float*)d_in[10];
  const float* kvl_w  = (const float*)d_in[11]; const float* kvl_b  = (const float*)d_in[12];
  const float* kvh_w  = (const float*)d_in[13]; const float* kvh_b  = (const float*)d_in[14];
  const float* ls_low = (const float*)d_in[15]; const float* ls_high= (const float*)d_in[16];
  const float* lconv_w= (const float*)d_in[17]; const float* lconv_b= (const float*)d_in[18];
  const float* hconv_w= (const float*)d_in[19]; const float* hconv_b= (const float*)d_in[20];
  const float* rconv_w= (const float*)d_in[21]; const float* rconv_b= (const float*)d_in[22];
  const float* fh_w   = (const float*)d_in[23]; const float* fh_b   = (const float*)d_in[24];
  const float* fa_w   = (const float*)d_in[25]; const float* fa_b   = (const float*)d_in[26];
  const float* proj_w = (const float*)d_in[27]; const float* proj_b = (const float*)d_in[28];
  const float* fc1_w  = (const float*)d_in[29]; const float* fc1_b  = (const float*)d_in[30];
  const float* fc2_w  = (const float*)d_in[31]; const float* fc2_b  = (const float*)d_in[32];

  // workspace carve-up (256B aligned)
  char* wp = (char*)d_ws;
  auto alloc = [&](size_t bytes) -> void* {
    void* r = (void*)wp;
    wp += (bytes + 255) & ~(size_t)255;
    return r;
  };
  const size_t NT = (size_t)MTOK * DIMC;      // 16.7M
  const size_t NH_ = (size_t)BATCH * 32 * 32 * DIMC;  // 4.19M
  float*  x1     = (float*) alloc(NT * 4);
  bf16_t* xw_bf  = (bf16_t*)alloc(NT * 2);
  bf16_t* wq_t   = (bf16_t*)alloc((size_t)256 * 256 * 2);
  bf16_t* wkvl_t = (bf16_t*)alloc((size_t)512 * 256 * 2);
  bf16_t* wkvh_t = (bf16_t*)alloc((size_t)512 * 256 * 2);
  bf16_t* wfh_t  = (bf16_t*)alloc((size_t)256 * 768 * 2);
  bf16_t* wfa_t  = (bf16_t*)alloc((size_t)256 * 768 * 2);
  bf16_t* wproj_t= (bf16_t*)alloc((size_t)256 * 256 * 2);
  bf16_t* wfc1_t = (bf16_t*)alloc((size_t)1024 * 256 * 2);
  bf16_t* wfc2_t = (bf16_t*)alloc((size_t)256 * 1024 * 2);
  float*  llF    = (float*) alloc(NH_ * 4);
  float*  lhF    = (float*) alloc(NH_ * 4);
  float*  hlF    = (float*) alloc(NH_ * 4);
  float*  hhF    = (float*) alloc(NH_ * 4);
  bf16_t* ll_bf  = (bf16_t*)alloc((size_t)MW16 * DIMC * 2);
  bf16_t* hi_bf  = (bf16_t*)alloc((size_t)MW16 * 768 * 2);
  float*  Qb     = (float*) alloc(NT * 4);
  float*  kvl    = (float*) alloc((size_t)MW16 * 512 * 4);
  float*  kvh    = (float*) alloc((size_t)MW16 * 512 * 4);
  bf16_t* xh_bf  = (bf16_t*)alloc((size_t)MW16 * DIMC * 2);
  bf16_t* cat    = (bf16_t*)alloc((size_t)MTOK * 768 * 2);
  float*  llc    = (float*) alloc(NH_ * 4);
  float*  lhc    = (float*) alloc(NH_ * 4);
  float*  hlc    = (float*) alloc(NH_ * 4);
  float*  hhc    = (float*) alloc(NH_ * 4);
  float*  idwtb  = (float*) alloc(NT * 4);
  float*  rstb   = (float*) alloc(NT * 4);
  bf16_t* alh_bf = (bf16_t*)alloc(NT * 2);
  float*  attn   = (float*) alloc(NT * 4);
  float*  x2     = (float*) alloc(NT * 4);
  float*  x3     = (float*) alloc(NT * 4);
  bf16_t* x3_bf  = (bf16_t*)alloc(NT * 2);
  bf16_t* hmid   = (bf16_t*)alloc((size_t)MTOK * 1024 * 2);
  float*  tbuf   = (float*) alloc(NT * 4);

  // 0) weights -> transposed bf16 (N x K)
  wconv_t_kernel<<<grid1(256*256),  256, 0, stream>>>(q_w,    wq_t,   256, 256);
  wconv_t_kernel<<<grid1(256*512),  256, 0, stream>>>(kvl_w,  wkvl_t, 256, 512);
  wconv_t_kernel<<<grid1(256*512),  256, 0, stream>>>(kvh_w,  wkvh_t, 256, 512);
  wconv_t_kernel<<<grid1(768*256),  256, 0, stream>>>(fh_w,   wfh_t,  768, 256);
  wconv_t_kernel<<<grid1(768*256),  256, 0, stream>>>(fa_w,   wfa_t,  768, 256);
  wconv_t_kernel<<<grid1(256*256),  256, 0, stream>>>(proj_w, wproj_t,256, 256);
  wconv_t_kernel<<<grid1(256*1024), 256, 0, stream>>>(fc1_w,  wfc1_t, 256, 1024);
  wconv_t_kernel<<<grid1(1024*256), 256, 0, stream>>>(fc2_w,  wfc2_t, 1024, 256);

  // 1) CPE0: x1 = x + dwconv(x)
  dwconv_kernel<0><<<grid1(NT), 256, 0, stream>>>(x, cpe0_w, cpe0_b, x1, BATCH, IMH, IMW);

  // 2) windowed bf16 gather for Q GEMM
  gather_win_kernel<<<grid1(NT), 256, 0, stream>>>(x1, xw_bf);

  // 3) Haar DWT (f32 half-res planes + windowed bf16 GEMM inputs)
  dwt_kernel<<<grid1(NH_), 256, 0, stream>>>(x1, llF, lhF, hlF, hhF, ll_bf, hi_bf);

  // 4) GEMMs: Q, kvl, xh=lrelu(fh), kvh
  launch_gemm(0, xw_bf, wq_t,   q_b,   Qb,    MTOK, 256, 256, stream);
  launch_gemm(0, ll_bf, wkvl_t, kvl_b, kvl,   MW16, 512, 256, stream);
  launch_gemm(3, hi_bf, wfh_t,  fh_b,  xh_bf, MW16, 256, 768, stream);
  launch_gemm(0, xh_bf, wkvh_t, kvh_b, kvh,   MW16, 512, 256, stream);

  // 5) attention (low + high) -> cat[:,0:256] and cat[:,256:512]
  attn_kernel<<<dim3(NWIN, NHEAD), 64, 0, stream>>>(Qb, kvl, ls_low,  cat, 0);
  attn_kernel<<<dim3(NWIN, NHEAD), 64, 0, stream>>>(Qb, kvh, ls_high, cat, 256);

  // 6) conv branch: lrelu(dwconv(..)) on sub-bands, idwt, rconv, scatter to cat[:,512:768]
  dwconv_kernel<1><<<grid1(NH_), 256, 0, stream>>>(llF, lconv_w, lconv_b, llc, BATCH, 32, 32);
  dwconv_kernel<1><<<grid1(NH_), 256, 0, stream>>>(lhF, hconv_w, hconv_b, lhc, BATCH, 32, 32);
  dwconv_kernel<1><<<grid1(NH_), 256, 0, stream>>>(hlF, hconv_w, hconv_b, hlc, BATCH, 32, 32);
  dwconv_kernel<1><<<grid1(NH_), 256, 0, stream>>>(hhF, hconv_w, hconv_b, hhc, BATCH, 32, 32);
  idwt_kernel<<<grid1(NT), 256, 0, stream>>>(llc, lhc, hlc, hhc, idwtb);
  dwconv_kernel<1><<<grid1(NT), 256, 0, stream>>>(idwtb, rconv_w, rconv_b, rstb, BATCH, IMH, IMW);
  scatter_rst_kernel<<<grid1(NT), 256, 0, stream>>>(rstb, cat);

  // 7) fuse + proj + LN + residual -> x2
  launch_gemm(1, cat,    wfa_t,   fa_b,   alh_bf, MTOK, 256, 768, stream);
  launch_gemm(0, alh_bf, wproj_t, proj_b, attn,   MTOK, 256, 256, stream);
  ln_add_kernel<1><<<MTOK, 256, 0, stream>>>(attn, x1, n1_g, n1_b, x2);

  // 8) CPE1: x3 = x2 + dwconv(x2); bf16 mirror for fc1
  dwconv_kernel<0><<<grid1(NT), 256, 0, stream>>>(x2, cpe1_w, cpe1_b, x3, BATCH, IMH, IMW);
  cvt_bf16_kernel<<<grid1(NT), 256, 0, stream>>>(x3, x3_bf, NT);

  // 9) MLP: h = gelu(x3@fc1), t = h@fc2, out = x3 + LN(t)
  launch_gemm(2, x3_bf, wfc1_t, fc1_b, hmid, MTOK, 1024, 256, stream);
  launch_gemm(0, hmid,  wfc2_t, fc2_b, tbuf, MTOK, 256, 1024, stream);
  ln_add_kernel<0><<<MTOK, 256, 0, stream>>>(tbuf, x3, n2_g, n2_b, (float*)d_out);
}
